// MABSINK_85744727098041
// MI455X (gfx1250) — matrix operand order repaired and checked by reading, not compiled, not run on hardware
//
#include <hip/hip_runtime.h>
#include <hip/hip_bf16.h>
#include <math.h>

#define DEVI __device__ __forceinline__

typedef __attribute__((ext_vector_type(16))) __bf16 v16bf;
typedef __attribute__((ext_vector_type(8)))  float  v8f;

union Frag16 {
  v16bf v;
  unsigned short u[16];
  uint4 q[2];
};

DEVI unsigned short f2bf(float f) {
  unsigned u = __builtin_bit_cast(unsigned, f);
  u += 0x7fffu + ((u >> 16) & 1u);   // round-to-nearest-even
  return (unsigned short)(u >> 16);
}

static constexpr int Bb = 4, Nn = 1024, Dd = 1024, DH = 128;
static constexpr int TM = 128, TN = 128, TK = 32;
static constexpr int SA = 40;                 // LDS row stride (halves), bank-conflict-free
static constexpr float LOG_MU = -6.9314616f;  // log(1/1024 + 1e-8)
static constexpr float SCALE_S = 0.03125f;    // 1/sqrt(1024)

// Load A fragment (16x32 bf16): lane = M row; K halves {0-7,16-23} / {8-15,24-31}
DEVI Frag16 load_a_frag(const unsigned short* base, int row, int lr, int hi) {
  Frag16 f;
  const unsigned short* rp = &base[(row + lr) * SA + hi * 8];
  f.q[0] = *(const uint4*)&rp[0];
  f.q[1] = *(const uint4*)&rp[16];
  return f;
}

// Load B fragment from transposed LDS tile (BT[n][k]): lane = N col; K halves {0-7,16-23} / {8-15,24-31}
DEVI Frag16 load_b_frag(const unsigned short* base, int col, int lr, int hi) {
  Frag16 f;
  const unsigned short* rp = &base[(col + lr) * SA + hi * 8];
  f.q[0] = *(const uint4*)&rp[0];
  f.q[1] = *(const uint4*)&rp[16];
  return f;
}

// ---------------------------------------------------------------------------
// Generic bf16-WMMA GEMM: C[4096,1024] = A[4096,1024] @ W[1024,1024] + bias
// MODE 0: k-proj  -> bf16 head-major [HB,N,DH]
// MODE 1: q-proj  -> bf16 head-major + f32 copy (for residual)
// MODE 2: v-proj  -> bf16 transposed [HB,DH,N]
// MODE 3: out-GEMM -> O2 = resid + relu(acc + bias)  (f32)
// ---------------------------------------------------------------------------
template<int MODE>
__global__ __launch_bounds__(256)
void gemm_proj(const float* __restrict__ A, const float* __restrict__ W,
               const float* __restrict__ bias,
               unsigned short* __restrict__ out_bf,
               float* __restrict__ out_f,
               const float* __restrict__ resid) {
  __shared__ __align__(16) unsigned short a_lds[TM * SA];
  __shared__ __align__(16) unsigned short b_lds[TN * SA];

  const int tid = threadIdx.x;
  const int lane = tid & 31, wv = tid >> 5;
  const int lr = lane & 15, hi = lane >> 4;
  const int n0 = blockIdx.x * TN;
  const int m0 = blockIdx.y * TM;

  v8f acc[8] = {};

  for (int k0 = 0; k0 < Dd; k0 += TK) {
    // speculative prefetch of next k-tile (global_prefetch_b8)
    if (k0 + TK < Dd) {
      __builtin_prefetch(&A[(size_t)(m0 + (tid >> 1)) * Dd + k0 + TK + (tid & 1) * 16], 0, 3);
      __builtin_prefetch(&W[(size_t)(k0 + TK + (tid >> 3)) * Dd + n0 + (tid & 7) * 16], 0, 3);
    }
    __syncthreads();
    // stage A tile 128x32 (f32 -> bf16)
    #pragma unroll
    for (int i = 0; i < 4; ++i) {
      int e = i * 1024 + tid * 4;
      int r = e >> 5, c = e & 31;
      const float4 av = *(const float4*)&A[(size_t)(m0 + r) * Dd + k0 + c];
      unsigned short* p = &a_lds[r * SA + c];
      p[0] = f2bf(av.x); p[1] = f2bf(av.y); p[2] = f2bf(av.z); p[3] = f2bf(av.w);
    }
    // stage W tile 32x128, transposed into LDS: b_lds[n][k]
    #pragma unroll
    for (int i = 0; i < 4; ++i) {
      int e = i * 1024 + tid * 4;
      int kk = e >> 7, nn = e & 127;
      const float4 wv4 = *(const float4*)&W[(size_t)(k0 + kk) * Dd + n0 + nn];
      b_lds[(nn + 0) * SA + kk] = f2bf(wv4.x);
      b_lds[(nn + 1) * SA + kk] = f2bf(wv4.y);
      b_lds[(nn + 2) * SA + kk] = f2bf(wv4.z);
      b_lds[(nn + 3) * SA + kk] = f2bf(wv4.w);
    }
    __syncthreads();

    const Frag16 af = load_a_frag(a_lds, wv * 16, lr, hi);
    Frag16 bfr[8];
    #pragma unroll
    for (int ct = 0; ct < 8; ++ct) bfr[ct] = load_b_frag(b_lds, ct * 16, lr, hi);
    #pragma unroll
    for (int ct = 0; ct < 8; ++ct)
      acc[ct] = __builtin_amdgcn_wmma_f32_16x16x32_bf16(
          false, af.v, false, bfr[ct].v, (short)0, acc[ct], false, false);
  }

  #pragma unroll
  for (int ct = 0; ct < 8; ++ct) {
    #pragma unroll
    for (int j = 0; j < 8; ++j) {
      const int gm = m0 + wv * 16 + hi * 8 + j;
      const int gn = n0 + ct * 16 + lr;
      float val = acc[ct][j] + bias[gn];
      if constexpr (MODE == 3) {
        const float r = resid[(size_t)gm * Dd + gn];
        out_f[(size_t)gm * Dd + gn] = r + fmaxf(val, 0.0f);
      } else {
        const int b = gm >> 10, nr = gm & 1023;
        const int h = gn >> 7,  d  = gn & 127;
        const int hb = h * Bb + b;
        if constexpr (MODE == 2) {
          out_bf[((size_t)hb * DH + d) * Nn + nr] = f2bf(val);
        } else {
          out_bf[((size_t)hb * Nn + nr) * DH + d] = f2bf(val);
          if constexpr (MODE == 1) out_f[(size_t)gm * Dd + gn] = val;
        }
      }
    }
  }
}

// ---------------------------------------------------------------------------
// S[hb] = q_[hb] (1024x128) @ k_[hb]^T (128x1024) * 1/sqrt(DV)
// ---------------------------------------------------------------------------
__global__ __launch_bounds__(256)
void sgemm_qk(const unsigned short* __restrict__ qb,
              const unsigned short* __restrict__ kb,
              float* __restrict__ S) {
  __shared__ __align__(16) unsigned short q_lds[TM * SA];
  __shared__ __align__(16) unsigned short k_lds[TN * SA];
  const int tid = threadIdx.x, lane = tid & 31, wv = tid >> 5;
  const int lr = lane & 15, hi = lane >> 4;
  const int nk0 = blockIdx.x * TN;
  const int nq0 = blockIdx.y * TM;
  const int hb  = blockIdx.z;

  v8f acc[8] = {};

  for (int d0 = 0; d0 < DH; d0 += TK) {
    if (d0 + TK < DH) {
      __builtin_prefetch(&qb[((size_t)hb * Nn + nq0 + (tid >> 1)) * DH + d0 + TK + (tid & 1) * 16], 0, 3);
      __builtin_prefetch(&kb[((size_t)hb * Nn + nk0 + (tid >> 1)) * DH + d0 + TK + (tid & 1) * 16], 0, 3);
    }
    __syncthreads();
    #pragma unroll
    for (int i = 0; i < 2; ++i) {
      int e = i * 2048 + tid * 8;
      int r = e >> 5, c = e & 31;
      *(uint4*)&q_lds[r * SA + c] =
          *(const uint4*)&qb[((size_t)hb * Nn + nq0 + r) * DH + d0 + c];
      *(uint4*)&k_lds[r * SA + c] =
          *(const uint4*)&kb[((size_t)hb * Nn + nk0 + r) * DH + d0 + c];
    }
    __syncthreads();

    const Frag16 af = load_a_frag(q_lds, wv * 16, lr, hi);
    Frag16 bfr[8];
    #pragma unroll
    for (int ct = 0; ct < 8; ++ct) bfr[ct] = load_b_frag(k_lds, ct * 16, lr, hi);
    #pragma unroll
    for (int ct = 0; ct < 8; ++ct)
      acc[ct] = __builtin_amdgcn_wmma_f32_16x16x32_bf16(
          false, af.v, false, bfr[ct].v, (short)0, acc[ct], false, false);
  }

  #pragma unroll
  for (int ct = 0; ct < 8; ++ct)
    #pragma unroll
    for (int j = 0; j < 8; ++j) {
      const int rr = nq0 + wv * 16 + hi * 8 + j;
      const int cc = nk0 + ct * 16 + lr;
      S[((size_t)hb * Nn + rr) * Nn + cc] = acc[ct][j] * SCALE_S;
    }
}

// ---------------------------------------------------------------------------
// u[row] = log_mu - logsumexp_m S[row, :]   (one wave per row, online LSE)
// ---------------------------------------------------------------------------
__global__ __launch_bounds__(256)
void row_lse(const float* __restrict__ S, float* __restrict__ u) {
  const int lane = threadIdx.x & 31, wv = threadIdx.x >> 5;
  const size_t row = (size_t)blockIdx.x * 8 + wv;
  const float* rp = &S[row * Nn];
  float mx = -3.0e38f, sm = 0.0f;
  for (int j = 0; j < 32; ++j) {
    const float v = rp[lane + j * 32];
    const float nm = fmaxf(mx, v);
    sm = sm * __expf(mx - nm) + __expf(v - nm);
    mx = nm;
  }
  for (int off = 16; off; off >>= 1) {
    const float om = __shfl_xor(mx, off, 32);
    const float os = __shfl_xor(sm, off, 32);
    const float nm = fmaxf(mx, om);
    sm = sm * __expf(mx - nm) + os * __expf(om - nm);
    mx = nm;
  }
  if (lane == 0) u[row] = LOG_MU - (mx + __logf(sm));
}

// ---------------------------------------------------------------------------
// vpot[hb,m] = log_nu - logsumexp_n (S[hb,n,m] + u[hb,n])  (thread per column)
// ---------------------------------------------------------------------------
__global__ __launch_bounds__(256)
void col_lse(const float* __restrict__ S, const float* __restrict__ u,
             float* __restrict__ vpot) {
  const int gc = blockIdx.x * 256 + threadIdx.x;
  const int hb = gc >> 10, m = gc & 1023;
  const float* base = &S[(size_t)hb * Nn * Nn + m];
  const float* ub = &u[(size_t)hb * Nn];
  float mx = -3.0e38f, sm = 0.0f;
  for (int n = 0; n < Nn; ++n) {
    const float v = base[(size_t)n * Nn] + ub[n];
    const float nm = fmaxf(mx, v);
    sm = sm * __expf(mx - nm) + __expf(v - nm);
    mx = nm;
  }
  vpot[gc] = LOG_MU - (mx + __logf(sm));   // log_nu == log_mu (NQ == NK)
}

// ---------------------------------------------------------------------------
// O0 = q + pi @ v_ ; pi tiles built on the fly: exp(S + u + vpot) -> bf16 LDS
// ---------------------------------------------------------------------------
__global__ __launch_bounds__(256)
void attn_pi_v(const float* __restrict__ S, const float* __restrict__ u,
               const float* __restrict__ vpot,
               const unsigned short* __restrict__ vbT,
               const float* __restrict__ qf, float* __restrict__ O0) {
  __shared__ __align__(16) unsigned short pi_lds[TM * SA];
  __shared__ __align__(16) unsigned short v_lds[TN * SA];
  const int tid = threadIdx.x, lane = tid & 31, wv = tid >> 5;
  const int lr = lane & 15, hi = lane >> 4;
  const int nq0 = blockIdx.x * TM;
  const int hb  = blockIdx.y;

  v8f acc[8] = {};

  for (int m0 = 0; m0 < Nn; m0 += TK) {
    if (m0 + TK < Nn) {
      __builtin_prefetch(&S[((size_t)hb * Nn + nq0 + (tid >> 1)) * Nn + m0 + TK + (tid & 1) * 16], 0, 3);
      __builtin_prefetch(&vbT[((size_t)hb * DH + (tid >> 1)) * Nn + m0 + TK + (tid & 1) * 16], 0, 3);
    }
    __syncthreads();
    #pragma unroll
    for (int i = 0; i < 4; ++i) {
      int e = i * 1024 + tid * 4;
      int r = e >> 5, c = e & 31;
      const float4 sv = *(const float4*)&S[((size_t)hb * Nn + nq0 + r) * Nn + m0 + c];
      const float4 vp = *(const float4*)&vpot[(size_t)hb * Nn + m0 + c];
      const float uu = u[(size_t)hb * Nn + nq0 + r];
      unsigned short* p = &pi_lds[r * SA + c];
      p[0] = f2bf(__expf(sv.x + uu + vp.x));
      p[1] = f2bf(__expf(sv.y + uu + vp.y));
      p[2] = f2bf(__expf(sv.z + uu + vp.z));
      p[3] = f2bf(__expf(sv.w + uu + vp.w));
    }
    #pragma unroll
    for (int i = 0; i < 2; ++i) {
      int e = i * 2048 + tid * 8;
      int d = e >> 5, c = e & 31;
      *(uint4*)&v_lds[d * SA + c] =
          *(const uint4*)&vbT[((size_t)hb * DH + d) * Nn + m0 + c];
    }
    __syncthreads();

    const Frag16 af = load_a_frag(pi_lds, wv * 16, lr, hi);
    Frag16 bfr[8];
    #pragma unroll
    for (int ct = 0; ct < 8; ++ct) bfr[ct] = load_b_frag(v_lds, ct * 16, lr, hi);
    #pragma unroll
    for (int ct = 0; ct < 8; ++ct)
      acc[ct] = __builtin_amdgcn_wmma_f32_16x16x32_bf16(
          false, af.v, false, bfr[ct].v, (short)0, acc[ct], false, false);
  }

  const int h = hb >> 2, b = hb & 3;
  #pragma unroll
  for (int ct = 0; ct < 8; ++ct)
    #pragma unroll
    for (int j = 0; j < 8; ++j) {
      const int n = nq0 + wv * 16 + hi * 8 + j;
      const int d = ct * 16 + lr;
      const size_t idx = ((size_t)b * Nn + n) * Dd + h * DH + d;
      O0[idx] = qf[idx] + acc[ct][j];
    }
}

// ---------------------------------------------------------------------------
// LayerNorm: one block per row of 1024
// ---------------------------------------------------------------------------
__global__ __launch_bounds__(256)
void layernorm(const float* __restrict__ in, const float* __restrict__ g,
               const float* __restrict__ be, float* __restrict__ out) {
  __shared__ float red[2][8];
  const int row = blockIdx.x, tid = threadIdx.x;
  const float4 x = *(const float4*)&in[(size_t)row * Dd + tid * 4];
  float s = x.x + x.y + x.z + x.w;
  float q = x.x * x.x + x.y * x.y + x.z * x.z + x.w * x.w;
  for (int off = 16; off; off >>= 1) {
    s += __shfl_xor(s, off, 32);
    q += __shfl_xor(q, off, 32);
  }
  if ((tid & 31) == 0) { red[0][tid >> 5] = s; red[1][tid >> 5] = q; }
  __syncthreads();
  s = 0.0f; q = 0.0f;
  #pragma unroll
  for (int i = 0; i < 8; ++i) { s += red[0][i]; q += red[1][i]; }
  const float mean = s * (1.0f / 1024.0f);
  const float var  = q * (1.0f / 1024.0f) - mean * mean;
  const float rstd = rsqrtf(var + 1e-5f);
  const float4 gg = *(const float4*)&g[tid * 4];
  const float4 bb = *(const float4*)&be[tid * 4];
  float4 o;
  o.x = (x.x - mean) * rstd * gg.x + bb.x;
  o.y = (x.y - mean) * rstd * gg.y + bb.y;
  o.z = (x.z - mean) * rstd * gg.z + bb.z;
  o.w = (x.w - mean) * rstd * gg.w + bb.w;
  *(float4*)&out[(size_t)row * Dd + tid * 4] = o;
}

// ---------------------------------------------------------------------------
extern "C" void kernel_launch(void* const* d_in, const int* in_sizes, int n_in,
                              void* d_out, int out_size, void* d_ws, size_t ws_size,
                              hipStream_t stream) {
  (void)in_sizes; (void)n_in; (void)out_size; (void)ws_size;
  const float* Q   = (const float*)d_in[0];
  const float* K   = (const float*)d_in[1];
  const float* Wq  = (const float*)d_in[2];
  const float* bq  = (const float*)d_in[3];
  const float* Wk  = (const float*)d_in[4];
  const float* bk  = (const float*)d_in[5];
  const float* Wv  = (const float*)d_in[6];
  const float* bv  = (const float*)d_in[7];
  const float* Wo  = (const float*)d_in[8];
  const float* bo  = (const float*)d_in[9];
  const float* g0  = (const float*)d_in[10];
  const float* be0 = (const float*)d_in[11];
  const float* g1  = (const float*)d_in[12];
  const float* be1 = (const float*)d_in[13];
  float* Out = (float*)d_out;

  char* ws = (char*)d_ws;
  const size_t MiB = (size_t)1 << 20;
  unsigned short* qb  = (unsigned short*)(ws + 0);        //  8 MiB bf16 [HB,N,DH]
  unsigned short* kb  = (unsigned short*)(ws + 8 * MiB);  //  8 MiB bf16 [HB,N,DH]
  unsigned short* vbT = (unsigned short*)(ws + 16 * MiB); //  8 MiB bf16 [HB,DH,N]
  float* qf   = (float*)(ws + 24 * MiB);                  // 16 MiB f32  [B,N,D]
  float* S    = (float*)(ws + 40 * MiB);                  // 128 MiB f32 [HB,N,N]
  float* u    = (float*)(ws + 168 * MiB);                 // 128 KiB
  float* vpot = (float*)(ws + 168 * MiB + 131072);        // 128 KiB
  float* O0   = (float*)(ws + 169 * MiB);                 // 16 MiB
  float* Oln  = (float*)(ws + 0);                         // reuse qb/kb (dead)
  float* O2   = (float*)(ws + 16 * MiB);                  // reuse vbT/qf (dead)

  const dim3 blk(256);
  gemm_proj<1><<<dim3(8, 32), blk, 0, stream>>>(Q, Wq, bq, qb, qf, nullptr);
  gemm_proj<0><<<dim3(8, 32), blk, 0, stream>>>(K, Wk, bk, kb, nullptr, nullptr);
  gemm_proj<2><<<dim3(8, 32), blk, 0, stream>>>(K, Wv, bv, vbT, nullptr, nullptr);
  sgemm_qk<<<dim3(8, 8, 32), blk, 0, stream>>>(qb, kb, S);
  row_lse<<<dim3(4096), blk, 0, stream>>>(S, u);
  col_lse<<<dim3(128), blk, 0, stream>>>(S, u, vpot);
  attn_pi_v<<<dim3(8, 32), blk, 0, stream>>>(S, u, vpot, vbT, qf, O0);
  layernorm<<<dim3(4096), blk, 0, stream>>>(O0, g0, be0, Oln);
  gemm_proj<3><<<dim3(8, 32), blk, 0, stream>>>(Oln, Wo, bo, nullptr, O2, Oln);
  layernorm<<<dim3(4096), blk, 0, stream>>>(O2, g1, be1, Out);
}
